// SMPNN_85341000171719
// MI455X (gfx1250) — matrix operand order repaired and checked by reading, not compiled, not run on hardware
//
#include <hip/hip_runtime.h>

// ---------------------------------------------------------------------------
// SMPNN on gfx1250: bf16 WMMA everywhere (node GEMM, fused edge-MLP, pooled
// readout). B operands pre-swizzled to WMMA-fragment order; A fragments are
// contiguous 2x16B loads; weight staging uses GLOBAL_LOAD_ASYNC_TO_LDS;
// native v_cvt bf16 conversion; sizes are compile-time multiples of 16 ->
// all tiles full, no per-element guards.
// N=100000, E=400000, H=128, G=50 (padded 64), L=6, NG=1024.
// ---------------------------------------------------------------------------

typedef __attribute__((ext_vector_type(16))) __bf16 v16bf;
typedef __attribute__((ext_vector_type(8)))  float  v8f;

union BF16x16 { v16bf v; unsigned short u[16]; uint4 q[2]; };

// Native hardware convert (RNE) -> v_cvt, 1 VALU op.
__device__ __forceinline__ unsigned short f2bf(float f) {
    __bf16 h = (__bf16)f;
    return __builtin_bit_cast(unsigned short, h);
}

__device__ __forceinline__ v8f wmma_bf16(v16bf a, v16bf b, v8f c) {
    return __builtin_amdgcn_wmma_f32_16x16x32_bf16(
        false, a, false, b, (short)0, c, false, false);
}

// Async 16B copy global -> LDS (ASYNCcnt-tracked, no VGPR round trip).
__device__ __forceinline__ void async_copy16(const uint4* g, uint4* l) {
    unsigned lo = (unsigned)(unsigned long long)l;
    asm volatile("global_load_async_to_lds_b128 %0, %1, off"
                 :: "v"(lo), "v"(g) : "memory");
}
__device__ __forceinline__ void async_wait0() {
    asm volatile("s_wait_asynccnt 0" ::: "memory");
}

// Packed-B offset for element (k,n), ncols-tile count NT = ncols/16.
__device__ __forceinline__ int packed_b_off(int k, int n, int NT) {
    int frag = (k >> 5) * NT + (n >> 4);
    int lane = ((k >> 3) & 1) * 16 + (n & 15);
    return (frag * 32 + lane) * 16 + ((k >> 4) & 1) * 8 + (k & 7);
}

// Load one packed fragment: 16 contiguous ushorts per lane.
__device__ __forceinline__ v16bf load_frag_packed(const unsigned short* base,
                                                  int frag, int lane) {
    BF16x16 f;
    const unsigned short* p = base + (frag * 32 + lane) * 16;
    f.q[0] = *(const uint4*)(p);
    f.q[1] = *(const uint4*)(p + 8);
    return f.v;
}

// A fragment from row-major bf16 [16, stride] (LDS or global).
__device__ __forceinline__ v16bf load_a_rowmajor(const unsigned short* base,
                                                 int stride, int kbase, int lane) {
    BF16x16 f;
    int m = lane & 15, hi = lane >> 4;
    const unsigned short* p = base + m * stride + kbase + hi * 8;
    f.q[0] = *(const uint4*)(p);
    f.q[1] = *(const uint4*)(p + 16);
    return f.v;
}

// ---------------------------------------------------------------------------
// Setup kernels
// ---------------------------------------------------------------------------

__global__ void pack_b128_kernel(const float* __restrict__ src,
                                 unsigned short* __restrict__ dst, int total) {
    int t = blockIdx.x * blockDim.x + threadIdx.x;
    if (t >= total) return;
    int n = t & 127, k = (t >> 7) & 127, l = t >> 14;
    dst[l * 128 * 128 + packed_b_off(k, n, 8)] = f2bf(src[t]);
}

__global__ void pack_w1_kernel(const float* __restrict__ src,
                               unsigned short* __restrict__ dst, int total) {
    int t = blockIdx.x * blockDim.x + threadIdx.x;
    if (t >= total) return;
    int n = t & 127, k = (t >> 7) & 63, l = t >> 13;
    float v = (k < 50) ? src[(l * 50 + k) * 128 + n] : 0.f;
    dst[l * 64 * 128 + packed_b_off(k, n, 8)] = f2bf(v);
}

__global__ void pack_pw1_kernel(const float* __restrict__ src,
                                unsigned short* __restrict__ dst, int total) {
    int t = blockIdx.x * blockDim.x + threadIdx.x;
    if (t >= total) return;
    int n = t & 63, k = t >> 6;
    dst[packed_b_off(k, n, 4)] = f2bf(src[t]);
}

__global__ void init_v0_kernel(const int* __restrict__ z0,
                               const float* __restrict__ emb,
                               unsigned short* __restrict__ v0bf,
                               float* __restrict__ acc, int total) {
    int t = blockIdx.x * blockDim.x + threadIdx.x;
    if (t >= total) return;
    int c = t & 127, n = t >> 7;
    v0bf[t] = f2bf(emb[z0[n] * 128 + c]);
    acc[t]  = 0.f;
}

__global__ void init_edges_kernel(const float* __restrict__ z1,
                                  const float* __restrict__ lw,
                                  const float* __restrict__ lb,
                                  unsigned short* __restrict__ v1bf, int total) {
    int t = blockIdx.x * blockDim.x + threadIdx.x;
    if (t >= total) return;
    int g = t & 63, e = t >> 6;
    float v = (g < 50) ? (z1[e] * lw[g] + lb[g]) : 0.f;
    v1bf[t] = f2bf(v);
}

__global__ void zero_out_kernel(float* __restrict__ out, int n) {
    int t = blockIdx.x * blockDim.x + threadIdx.x;
    if (t < n) out[t] = 0.f;
}

// ---------------------------------------------------------------------------
// Per-layer kernels
// ---------------------------------------------------------------------------

// x = v0 @ lin_w[l].  A direct from global bf16; B packed in LDS (async).
__global__ __launch_bounds__(256) void node_gemm_kernel(
    const unsigned short* __restrict__ v0bf,
    const unsigned short* __restrict__ wpk,     // packed [128,128]
    float* __restrict__ x, int nrows) {
    __shared__ __align__(16) unsigned short w_lds[128 * 128];
    int tid = threadIdx.x, lane = tid & 31, wave = tid >> 5;
    {
        uint4* d = (uint4*)w_lds;
        const uint4* s = (const uint4*)wpk;
        for (int t = tid; t < 128 * 128 / 8; t += 256) async_copy16(s + t, d + t);
    }
    async_wait0();
    __syncthreads();

    int tile = blockIdx.x * 8 + wave;
    int rowbase = tile * 16;
    if (rowbase >= nrows) return;                 // wave-uniform; tiles are full

    int n = lane & 15, hi = lane >> 4;
    const unsigned short* rp = v0bf + (size_t)(rowbase + n) * 128;

    v16bf afr[4];
#pragma unroll
    for (int ks = 0; ks < 4; ++ks) {
        BF16x16 f;
        const unsigned short* p = rp + ks * 32 + hi * 8;
        f.q[0] = *(const uint4*)(p);
        f.q[1] = *(const uint4*)(p + 16);
        afr[ks] = f.v;
    }

    v16bf bfr = load_frag_packed(w_lds, 0, lane);   // (nt=0,ks=0)
#pragma unroll
    for (int nt = 0; nt < 8; ++nt) {
        v8f acc = {};
#pragma unroll
        for (int ks = 0; ks < 4; ++ks) {
            v16bf bcur = bfr;
            int idx = nt * 4 + ks + 1;              // next (nt',ks')
            if (idx < 32)
                bfr = load_frag_packed(w_lds, (idx & 3) * 8 + (idx >> 2), lane);
            acc = wmma_bf16(afr[ks], bcur, acc);
        }
        float* xp = x + (size_t)(rowbase + 8 * hi) * 128 + nt * 16 + n;
#pragma unroll
        for (int r = 0; r < 8; ++r) xp[r * 128] = acc[r];   // clause, imm offsets
    }
}

// Fused: h = relu(v1 @ w1p + b1) @ w2 + b2; acc[i] += x[j]*h ; acc[j] += x[i]*h.
__global__ __launch_bounds__(256) void edge_mlp_scatter_kernel(
    const unsigned short* __restrict__ v1bf,   // [E,64] bf16 row-major
    const unsigned short* __restrict__ w1pk,   // packed [64,128]
    const float* __restrict__ b1,
    const unsigned short* __restrict__ w2pk,   // packed [128,128]
    const float* __restrict__ b2,
    const float* __restrict__ x,               // [N,128]
    const int* __restrict__ ei,                // [2,E] flat
    float* __restrict__ acc,                   // [N,128]
    int nE) {
    __shared__ __align__(16) unsigned short w1_lds[64 * 128];
    __shared__ __align__(16) unsigned short w2_lds[128 * 128];
    __shared__ __align__(16) unsigned short mid_lds[8][16 * 128]; // A1 rows -> A2 packed
    __shared__ int idx_lds[8][32];

    int tid = threadIdx.x, lane = tid & 31, wave = tid >> 5;
    {
        uint4* d1 = (uint4*)w1_lds;  const uint4* s1 = (const uint4*)w1pk;
        for (int t = tid; t < 64 * 128 / 8; t += 256) async_copy16(s1 + t, d1 + t);
        uint4* d2 = (uint4*)w2_lds;  const uint4* s2 = (const uint4*)w2pk;
        for (int t = tid; t < 128 * 128 / 8; t += 256) async_copy16(s2 + t, d2 + t);
    }

    int tile = blockIdx.x * 8 + wave;
    int eb = tile * 16;
    bool active = eb < nE;                        // wave-uniform; tiles are full
    unsigned short* mlw = &mid_lds[wave][0];
    int n = lane & 15, hi = lane >> 4;

    float b1v[8], b2v[8];
    if (active) {
        uint4* dm = (uint4*)mlw;
        const uint4* sv = (const uint4*)(v1bf + (size_t)eb * 64);
        for (int t = lane; t < 16 * 64 / 8; t += 32) async_copy16(sv + t, dm + t);
        if (lane < 16) {
            idx_lds[wave][lane]      = ei[eb + lane];        // edge_index0[0][e]
            idx_lds[wave][16 + lane] = ei[nE + eb + lane];   // edge_index0[1][e]
        }
        // biases into registers (LOADcnt, keeps DScnt clean for B pipeline)
#pragma unroll
        for (int nt = 0; nt < 8; ++nt) {
            b1v[nt] = b1[nt * 16 + n];
            b2v[nt] = b2[nt * 16 + n];
        }
    }
    async_wait0();
    __syncthreads();
    if (!active) return;

    // --- GEMM1: [16x64] @ [64x128]; relu; store mid in A-fragment-packed LDS ---
    v16bf a1_0 = load_a_rowmajor(mlw, 64, 0,  lane);
    v16bf a1_1 = load_a_rowmajor(mlw, 64, 32, lane);
    v16bf b1f = load_frag_packed(w1_lds, 0, lane);
#pragma unroll
    for (int nt = 0; nt < 8; ++nt) {
        v8f c1 = {};
#pragma unroll
        for (int ks = 0; ks < 2; ++ks) {
            v16bf bcur = b1f;
            int idx = nt * 2 + ks + 1;
            if (idx < 16)
                b1f = load_frag_packed(w1_lds, (idx & 1) * 8 + (idx >> 1), lane);
            c1 = wmma_bf16(ks == 0 ? a1_0 : a1_1, bcur, c1);
        }
        float bb = b1v[nt];
        // static C->A permutation: GEMM2 k = col
        int kstep  = nt >> 1;
        int klocal = (nt & 1) * 16 + n;
        int hi2 = (klocal >> 3) & 1;
        int e   = (klocal < 16) ? (klocal & 7) : (8 + (klocal & 7));
        unsigned short* mp = mlw + kstep * 512 + (8 * hi + 16 * hi2) * 16 + e;
#pragma unroll
        for (int r = 0; r < 8; ++r) {
            float hv = c1[r] + bb;
            hv = hv > 0.f ? hv : 0.f;
            mp[r * 16] = f2bf(hv);                 // v_cvt + ds_store_b16
        }
    }
    asm volatile("s_wait_dscnt 0" ::: "memory");   // LDS round-trip (same wave)

    // --- GEMM2 + gather/gate/scatter ---
    v16bf a2[4];
#pragma unroll
    for (int ks = 0; ks < 4; ++ks)
        a2[ks] = load_frag_packed(mlw, ks, lane);

    // hoisted per-row element offsets (fit in u32: N*128 = 12.8M)
    unsigned ioff[8], joff[8];
#pragma unroll
    for (int r = 0; r < 8; ++r) {
        int m = r + 8 * hi;
        ioff[r] = (unsigned)idx_lds[wave][m]      * 128u + (unsigned)n;
        joff[r] = (unsigned)idx_lds[wave][16 + m] * 128u + (unsigned)n;
    }

    v16bf b2f = load_frag_packed(w2_lds, 0, lane);
#pragma unroll
    for (int nt = 0; nt < 8; ++nt) {
        v8f c2 = {};
#pragma unroll
        for (int ks = 0; ks < 4; ++ks) {
            v16bf bcur = b2f;
            int idx = nt * 4 + ks + 1;
            if (idx < 32)
                b2f = load_frag_packed(w2_lds, (idx & 3) * 8 + (idx >> 2), lane);
            c2 = wmma_bf16(a2[ks], bcur, c2);
        }
        float bb = b2v[nt];
#pragma unroll
        for (int r = 0; r < 8; ++r) {
            float h = c2[r] + bb;
            float xj = x[joff[r] + nt * 16];       // L2-resident gathers
            float xi = x[ioff[r] + nt * 16];
            atomicAdd(&acc[ioff[r] + nt * 16], xj * h);
            atomicAdd(&acc[joff[r] + nt * 16], xi * h);
        }
    }
}

// v0bf = bf16(relu(acc + bias)); acc = 0 for next layer
__global__ void bias_relu_kernel(unsigned short* __restrict__ v0bf,
                                 float* __restrict__ acc,
                                 const float* __restrict__ bias, int total) {
    int t = blockIdx.x * blockDim.x + threadIdx.x;
    if (t >= total) return;
    float v = acc[t] + bias[t & 127];
    v = v > 0.f ? v : 0.f;
    v0bf[t] = f2bf(v);
    acc[t]  = 0.f;
}

// ---------------------------------------------------------------------------
// Readout via WMMA: t = softplus(v0 @ pw1 + pb1) - ln2; s = t @ pw2 + pb2;
// segment_sum over batch.  16 nodes per wave.
// ---------------------------------------------------------------------------
__global__ __launch_bounds__(256) void pool_kernel(
    const unsigned short* __restrict__ v0bf,
    const unsigned short* __restrict__ pw1pk,   // packed [128,64]
    const float* __restrict__ pb1, const float* __restrict__ pw2,
    const float* __restrict__ pb2, const int* __restrict__ batch,
    float* __restrict__ out, int nrows) {
    __shared__ __align__(16) unsigned short w_lds[128 * 64];
    int tid = threadIdx.x, lane = tid & 31, wave = tid >> 5;
    {
        uint4* d = (uint4*)w_lds;
        const uint4* s = (const uint4*)pw1pk;
        for (int t = tid; t < 128 * 64 / 8; t += 256) async_copy16(s + t, d + t);
    }
    async_wait0();
    __syncthreads();

    int tile = blockIdx.x * 8 + wave;
    int rowbase = tile * 16;
    if (rowbase >= nrows) return;                 // wave-uniform; tiles are full

    int n = lane & 15, hi = lane >> 4;
    const unsigned short* rp = v0bf + (size_t)(rowbase + n) * 128;

    v16bf afr[4];
#pragma unroll
    for (int ks = 0; ks < 4; ++ks) {
        BF16x16 f;
        const unsigned short* p = rp + ks * 32 + hi * 8;
        f.q[0] = *(const uint4*)(p);
        f.q[1] = *(const uint4*)(p + 16);
        afr[ks] = f.v;
    }

    float contrib[8];
#pragma unroll
    for (int r = 0; r < 8; ++r) contrib[r] = 0.f;

    v16bf bfr = load_frag_packed(w_lds, 0, lane);
#pragma unroll
    for (int nt = 0; nt < 4; ++nt) {
        v8f c = {};
#pragma unroll
        for (int ks = 0; ks < 4; ++ks) {
            v16bf bcur = bfr;
            int idx = nt * 4 + ks + 1;
            if (idx < 16)
                bfr = load_frag_packed(w_lds, (idx & 3) * 4 + (idx >> 2), lane);
            c = wmma_bf16(afr[ks], bcur, c);
        }
        int col = nt * 16 + n;
        float bb = pb1[col];
        float w2v = pw2[col];
#pragma unroll
        for (int r = 0; r < 8; ++r) {
            float s = c[r] + bb;
            float sp = (s > 20.f) ? s : log1pf(expf(s));
            contrib[r] += (sp - 0.6931471805599453f) * w2v;   // ShiftedSoftplus
        }
    }
    // reduce over the 16 lanes of each half-wave (channels)
#pragma unroll
    for (int off = 1; off < 16; off <<= 1)
#pragma unroll
        for (int r = 0; r < 8; ++r)
            contrib[r] += __shfl_xor(contrib[r], off, 32);

    if (n == 0) {
        float b2v = pb2[0];
#pragma unroll
        for (int r = 0; r < 8; ++r)
            atomicAdd(&out[batch[rowbase + r + 8 * hi]], contrib[r] + b2v);
    }
}

// ---------------------------------------------------------------------------
extern "C" void kernel_launch(void* const* d_in, const int* in_sizes, int n_in,
                              void* d_out, int out_size, void* d_ws, size_t ws_size,
                              hipStream_t stream) {
    const int N = 100000, E = 400000, H = 128, L = 6, NG = 1024;

    const int*   z0     = (const int*)d_in[0];
    const float* z1     = (const float*)d_in[1];
    const int*   ei     = (const int*)d_in[2];
    const int*   batch  = (const int*)d_in[3];
    const float* emb    = (const float*)d_in[4];
    const float* lin1_w = (const float*)d_in[5];
    const float* lin1_b = (const float*)d_in[6];
    const float* lin_w  = (const float*)d_in[7];
    const float* bias   = (const float*)d_in[8];
    const float* mlp_w1 = (const float*)d_in[9];
    const float* mlp_b1 = (const float*)d_in[10];
    const float* mlp_w2 = (const float*)d_in[11];
    const float* mlp_b2 = (const float*)d_in[12];
    const float* pw1    = (const float*)d_in[13];
    const float* pb1    = (const float*)d_in[14];
    const float* pw2    = (const float*)d_in[15];
    const float* pb2    = (const float*)d_in[16];
    float* out = (float*)d_out;

    // workspace carve-up
    char* ws = (char*)d_ws;
    float* x   = (float*)ws;                      ws += (size_t)N * H * 4;
    float* acc = (float*)ws;                      ws += (size_t)N * H * 4;
    unsigned short* v0bf  = (unsigned short*)ws;  ws += (size_t)N * H * 2;
    unsigned short* v1bf  = (unsigned short*)ws;  ws += (size_t)E * 64 * 2;
    unsigned short* wlin  = (unsigned short*)ws;  ws += (size_t)L * H * H * 2;
    unsigned short* w1p   = (unsigned short*)ws;  ws += (size_t)L * 64 * H * 2;
    unsigned short* w2    = (unsigned short*)ws;  ws += (size_t)L * H * H * 2;
    unsigned short* pw1pk = (unsigned short*)ws;  ws += (size_t)H * 64 * 2;

    // setup
    {
        int n = L * H * H;
        pack_b128_kernel<<<(n + 255) / 256, 256, 0, stream>>>(lin_w, wlin, n);
        pack_b128_kernel<<<(n + 255) / 256, 256, 0, stream>>>(mlp_w2, w2, n);
        int np = L * 64 * H;
        pack_w1_kernel<<<(np + 255) / 256, 256, 0, stream>>>(mlp_w1, w1p, np);
        int npw = H * 64;
        pack_pw1_kernel<<<(npw + 255) / 256, 256, 0, stream>>>(pw1, pw1pk, npw);
        int nv = N * H;
        init_v0_kernel<<<(nv + 255) / 256, 256, 0, stream>>>(z0, emb, v0bf, acc, nv);
        int ne = E * 64;
        init_edges_kernel<<<(ne + 255) / 256, 256, 0, stream>>>(z1, lin1_w, lin1_b, v1bf, ne);
    }

    int node_blocks = ((N + 15) / 16 + 7) / 8;
    int edge_blocks = ((E + 15) / 16 + 7) / 8;

    for (int l = 0; l < L; ++l) {
        node_gemm_kernel<<<node_blocks, 256, 0, stream>>>(
            v0bf, wlin + (size_t)l * H * H, x, N);
        edge_mlp_scatter_kernel<<<edge_blocks, 256, 0, stream>>>(
            v1bf, w1p + (size_t)l * 64 * H, mlp_b1 + (size_t)l * H,
            w2 + (size_t)l * H * H, mlp_b2 + (size_t)l * H,
            x, ei, acc, E);
        bias_relu_kernel<<<(N * H + 255) / 256, 256, 0, stream>>>(
            v0bf, acc, bias + (size_t)l * H, N * H);
    }

    zero_out_kernel<<<(NG + 255) / 256, 256, 0, stream>>>(out, NG);
    pool_kernel<<<node_blocks, 256, 0, stream>>>(
        v0bf, pw1pk, pb1, pw2, pb2, batch, out, N);
}
